// GeoDavidCollective_51797305589996
// MI455X (gfx1250) — compile-verified, matrix operand order checked by reading, not run on hardware
//
#include <hip/hip_runtime.h>
#include <math.h>

// ---------------------------------------------------------------------------
// CDNA5 WMMA types (wave32, v_wmma_f32_16x16x32_bf16)
// ---------------------------------------------------------------------------
typedef __attribute__((ext_vector_type(16))) __bf16 v16bf;
typedef __attribute__((ext_vector_type(8)))  float  v8f;
typedef unsigned short u16;
typedef unsigned int   u32;

#define B_ROWS      8192
#define INPUT_DIM   1280
#define SCALE_DIM   512
#define BELLY_DIM   1024
#define NUM_BINS    100
#define NUM_BINS_P  128      // Wt rows padded (zero-filled) to multiple of 128
#define TOTAL_PAT   1000
#define TOTAL_PAT_P 1024     // Wp rows padded (zero-filled) to multiple of 128

// RNE float -> bf16 bits
__device__ __forceinline__ u32 rne_bits(float f) {
    union { float f; u32 u; } v; v.f = f;
    return v.u + 0x7FFFu + ((v.u >> 16) & 1u);
}
__device__ __forceinline__ u16 f2bf_u16(float f) {
    return (u16)(rne_bits(f) >> 16);
}
__device__ __forceinline__ u32 pack_bf(u32 rhi, u32 rlo) {
    return (rhi & 0xFFFF0000u) | (rlo >> 16);
}

// ---------------------------------------------------------------------------
// One-shot fp32 -> packed bf16 conversion (bandwidth-bound, outside hot loop)
// ---------------------------------------------------------------------------
__global__ __launch_bounds__(256)
void cvt_f32_to_bf16(const float* __restrict__ in, u16* __restrict__ out, long long n)
{
    long long i = 4LL * ((long long)blockIdx.x * blockDim.x + threadIdx.x);
    if (i >= n) return;
    float4 f = *(const float4*)(in + i);
    u32 lo = pack_bf(rne_bits(f.y), rne_bits(f.x));
    u32 hi = pack_bf(rne_bits(f.w), rne_bits(f.z));
    *(uint2*)(out + i) = make_uint2(lo, hi);
}

// Zero-fill a u16 region (for weight row padding); n multiple of 8.
__global__ __launch_bounds__(256)
void fill_zero_u16(u16* __restrict__ p, long long n)
{
    long long i = 8LL * ((long long)blockIdx.x * blockDim.x + threadIdx.x);
    if (i >= n) return;
    *(uint4*)(p + i) = make_uint4(0u, 0u, 0u, 0u);
}

// ---------------------------------------------------------------------------
// bf16 fragment load (16x32, A layout; B identical on weight rows since
// Y = X * W^T).  lane = half*16 + r holds row r; element e maps to
// K = k0 + (e<8?0:16) + 8*half + (e&7) -> two contiguous b128 loads.
// Unconditional (operands are padded so every row is valid).
// ---------------------------------------------------------------------------
__device__ __forceinline__ v16bf load_frag_bf(const u16* __restrict__ p)
{
    struct F2 { uint4 a, b; } t;
    t.a = *(const uint4*)(p);        // elements 0..7
    t.b = *(const uint4*)(p + 16);   // elements 8..15
    __builtin_prefetch(p + 32, 0, 1);   // next K-slab -> global_prefetch_b8
    return __builtin_bit_cast(v16bf, t);
}

// ---------------------------------------------------------------------------
// GEMM: Y[M,Nout] = act(X[M,K](bf16) * W[Npad,K](bf16)^T + bias[Nout])
// Block = 256 thr (8 waves) -> 128x128 tile; wave tile 32x64 = 2x4 accums ->
// 8 v_wmma_f32_16x16x32_bf16 per K-step vs 12 unconditional b128 loads.
// M % 128 == 0, K % 32 == 0, grid.x*128 <= Npad (weights zero-padded).
// Stores predicated on n < Nout only in the epilogue.
// ---------------------------------------------------------------------------
template<int RELU, int OUTF, int OUTH>
__global__ __launch_bounds__(256)
void gemm_wmma_bf16(const u16* __restrict__ X, const u16* __restrict__ W,
                    const float* __restrict__ bias,
                    float* __restrict__ Yf, u16* __restrict__ Yh,
                    int M, int Nout, int K)
{
    const int lane  = threadIdx.x & 31;
    const int wave  = threadIdx.x >> 5;   // 0..7
    const int half  = lane >> 4;          // 0/1
    const int r     = lane & 15;
    const int waveM = wave & 3;           // 0..3
    const int waveN = wave >> 2;          // 0..1
    const int m0    = blockIdx.y * 128 + waveM * 32;
    const int n0    = blockIdx.x * 128 + waveN * 64;

    v8f acc[2][4] = {};

    const u16* pa0 = X + (long long)(m0 + r)      * K + half * 8;
    const u16* pa1 = X + (long long)(m0 + 16 + r) * K + half * 8;
    const u16* pb0 = W + (long long)(n0 + r)      * K + half * 8;
    const u16* pb1 = W + (long long)(n0 + 16 + r) * K + half * 8;
    const u16* pb2 = W + (long long)(n0 + 32 + r) * K + half * 8;
    const u16* pb3 = W + (long long)(n0 + 48 + r) * K + half * 8;

    for (int k0 = 0; k0 < K; k0 += 32) {
        v16bf aF[2], bF[4];
        aF[0] = load_frag_bf(pa0 + k0);
        aF[1] = load_frag_bf(pa1 + k0);
        bF[0] = load_frag_bf(pb0 + k0);
        bF[1] = load_frag_bf(pb1 + k0);
        bF[2] = load_frag_bf(pb2 + k0);
        bF[3] = load_frag_bf(pb3 + k0);
        #pragma unroll
        for (int mi = 0; mi < 2; ++mi)
            #pragma unroll
            for (int ni = 0; ni < 4; ++ni)
                acc[mi][ni] = __builtin_amdgcn_wmma_f32_16x16x32_bf16(
                    /*neg_a=*/false, aF[mi], /*neg_b=*/false, bF[ni],
                    /*c_mod=*/(short)0, acc[mi][ni],
                    /*reuse_a=*/false, /*reuse_b=*/false);
    }

    // C/D layout: lane -> N = r (+ tile), VGPR v -> M = v + 8*half
    #pragma unroll
    for (int mi = 0; mi < 2; ++mi) {
        #pragma unroll
        for (int ni = 0; ni < 4; ++ni) {
            int n = n0 + 16*ni + r;
            if (n < Nout) {
                float bv = bias ? bias[n] : 0.0f;
                #pragma unroll
                for (int v = 0; v < 8; ++v) {
                    int m = m0 + 16*mi + v + 8*half;
                    float val = acc[mi][ni][v] + bv;
                    if (RELU) val = fmaxf(val, 0.0f);
                    long long idx = (long long)m * Nout + n;
                    if (OUTF) Yf[idx] = val;
                    if (OUTH) Yh[idx] = f2bf_u16(val);
                }
            }
        }
    }
}

// ---------------------------------------------------------------------------
// Per-row: L2-normalize z (D=512) -> fp32 output + bf16 copy for head GEMMs,
// fused f2p dot -> shift, timestep class, atomicMax for max_pos.
// ---------------------------------------------------------------------------
__global__ __launch_bounds__(256)
void normalize_row(const float* __restrict__ zraw, const float* __restrict__ f2p_w,
                   const float* __restrict__ f2p_b, const float* __restrict__ timesteps,
                   float* __restrict__ z_out, u16* __restrict__ zbf_out,
                   float* __restrict__ shift_out,
                   int* __restrict__ cls_out, float* __restrict__ cls_out_f,
                   int* __restrict__ posmax)
{
    const int row = blockIdx.x;
    const int t   = threadIdx.x;
    const float* zr = zraw + (long long)row * SCALE_DIM;

    float v0 = zr[t], v1 = zr[t + 256];

    __shared__ float red[256];
    __shared__ float scale_sh;

    red[t] = v0 * v0 + v1 * v1;
    __syncthreads();
    for (int s = 128; s > 0; s >>= 1) {
        if (t < s) red[t] += red[t + s];
        __syncthreads();
    }
    if (t == 0) {
        float nrm = sqrtf(red[0]);
        scale_sh = 1.0f / fmaxf(nrm, 1e-12f);
    }
    __syncthreads();

    const float sc = scale_sh;
    float z0 = v0 * sc, z1 = v1 * sc;
    long long base = (long long)row * SCALE_DIM;
    z_out[base + t]         = z0;
    z_out[base + t + 256]   = z1;
    zbf_out[base + t]       = f2bf_u16(z0);
    zbf_out[base + t + 256] = f2bf_u16(z1);

    __syncthreads();
    red[t] = z0 * f2p_w[t] + z1 * f2p_w[t + 256];
    __syncthreads();
    for (int s = 128; s > 0; s >>= 1) {
        if (t < s) red[t] += red[t + s];
        __syncthreads();
    }
    if (t == 0) {
        shift_out[row] = tanhf(red[0] + f2p_b[0]) * 0.3f;
        float ts = timesteps[row];
        int c = (int)(ts * (1.0f / 1000.0f) * (float)NUM_BINS);  // trunc, per .astype(int32)
        c = min(max(c, 0), NUM_BINS - 1);
        cls_out[row]   = c;
        cls_out_f[row] = (float)c;
        atomicMax(posmax, c);
    }
}

// ---------------------------------------------------------------------------
// Cantor staircase: pure VALU, 12 fully unrolled levels.
// ---------------------------------------------------------------------------
__global__ __launch_bounds__(256)
void cantor_kernel(const int* __restrict__ cls, const float* __restrict__ shift,
                   const int* __restrict__ posmax, const float* __restrict__ alpha_p,
                   float* __restrict__ out, int n)
{
    int i = blockIdx.x * blockDim.x + threadIdx.x;
    if (i >= n) return;

    const float alpha  = alpha_p[0];
    const int   maxpos = posmax[0] + 1;
    const float denom  = (float)max(maxpos - 1, 1);
    const float posf   = (float)cls[i];

    float xb = (maxpos > 1) ? (posf / denom) : posf;
    xb = fminf(fmaxf(xb, 1e-6f), 1.0f - 1e-6f);
    float x = fminf(fmaxf(xb + shift[i], 1e-6f), 1.0f - 1e-6f);

    const float invtau = 1.0f / (0.25f + 1e-8f);
    float w = 0.5f, Cx = 0.0f;
    #pragma unroll
    for (int l = 0; l < 12; ++l) {
        float y  = x * 3.0f;
        float d0 = (y - 0.5f) * (y - 0.5f);
        float d1 = (y - 1.5f) * (y - 1.5f);
        float d2 = (y - 2.5f) * (y - 2.5f);
        float m  = fminf(d0, fminf(d1, d2));
        float e0 = __expf(-(d0 - m) * invtau);
        float e1 = __expf(-(d1 - m) * invtau);
        float e2 = __expf(-(d2 - m) * invtau);
        float inv = 1.0f / (e0 + e1 + e2);
        Cx += (e1 * inv * alpha + e2 * inv) * w;
        x = y - floorf(y);
        w *= 0.5f;
    }
    out[i] = fminf(fmaxf(Cx, 0.0f), 1.0f);
}

__global__ void init_scalar(int* p) { if (threadIdx.x == 0) *p = 0; }

// ---------------------------------------------------------------------------
// Pipeline: zero-pad + convert operands to bf16 -> GEMM1(relu, bf16 out) ->
// GEMM2(fp32 out) -> normalize -> head GEMMs -> cantor.  All on `stream`.
// ---------------------------------------------------------------------------
static inline size_t align256(size_t x) { return (x + 255) & ~(size_t)255; }

extern "C" void kernel_launch(void* const* d_in, const int* in_sizes, int n_in,
                              void* d_out, int out_size, void* d_ws, size_t ws_size,
                              hipStream_t stream) {
    (void)in_sizes; (void)n_in; (void)out_size; (void)ws_size;

    const float* features  = (const float*)d_in[0];
    const float* timesteps = (const float*)d_in[1];
    const float* W1        = (const float*)d_in[2];
    const float* b1        = (const float*)d_in[3];
    const float* W2        = (const float*)d_in[4];
    const float* Wt        = (const float*)d_in[5];
    const float* bt        = (const float*)d_in[6];
    const float* Wp        = (const float*)d_in[7];
    const float* bp        = (const float*)d_in[8];
    const float* f2p_w     = (const float*)d_in[9];
    const float* f2p_b     = (const float*)d_in[10];
    const float* alpha     = (const float*)d_in[11];

    // Output layout (flat, reference return order)
    float* out       = (float*)d_out;
    float* z_out     = out;                                      // [8192*512]
    float* tl_out    = z_out  + (size_t)B_ROWS * SCALE_DIM;      // [8192*100]
    float* pl_out    = tl_out + (size_t)B_ROWS * NUM_BINS;       // [8192*1000]
    float* cls_out_f = pl_out + (size_t)B_ROWS * TOTAL_PAT;      // [8192]
    float* cant_out  = cls_out_f + B_ROWS;                       // [8192]

    // Workspace layout (bf16 operands, weight rows zero-padded to 128)
    char* ws = (char*)d_ws;
    size_t off = 0;
    u16* feat_bf = (u16*)(ws + off); off = align256(off + (size_t)B_ROWS * INPUT_DIM * 2);
    u16* W1_bf   = (u16*)(ws + off); off = align256(off + (size_t)BELLY_DIM * INPUT_DIM * 2);
    u16* W2_bf   = (u16*)(ws + off); off = align256(off + (size_t)SCALE_DIM * BELLY_DIM * 2);
    u16* Wt_bf   = (u16*)(ws + off); off = align256(off + (size_t)NUM_BINS_P * SCALE_DIM * 2);
    u16* Wp_bf   = (u16*)(ws + off); off = align256(off + (size_t)TOTAL_PAT_P * SCALE_DIM * 2);
    u16* h_bf    = (u16*)(ws + off); off = align256(off + (size_t)B_ROWS * BELLY_DIM * 2);
    u16* z_bf    = (u16*)(ws + off); off = align256(off + (size_t)B_ROWS * SCALE_DIM * 2);
    float* zraw  = (float*)(ws + off); off = align256(off + (size_t)B_ROWS * SCALE_DIM * 4);
    float* shift = (float*)(ws + off); off = align256(off + (size_t)B_ROWS * 4);
    int*   cls   = (int*)(ws + off);   off = align256(off + (size_t)B_ROWS * 4);
    int*   pmax  = (int*)(ws + off);   off = align256(off + 256);  // prefetch slack after

    init_scalar<<<1, 1, 0, stream>>>(pmax);

    // Zero the padded tail rows of Wt/Wp (then conversion fills valid rows)
    {
        long long npadt = (long long)(NUM_BINS_P - NUM_BINS) * SCALE_DIM;   // 28*512
        long long npadp = (long long)(TOTAL_PAT_P - TOTAL_PAT) * SCALE_DIM; // 24*512
        fill_zero_u16<<<(unsigned)((npadt / 8 + 255) / 256), 256, 0, stream>>>(
            Wt_bf + (size_t)NUM_BINS * SCALE_DIM, npadt);
        fill_zero_u16<<<(unsigned)((npadp / 8 + 255) / 256), 256, 0, stream>>>(
            Wp_bf + (size_t)TOTAL_PAT * SCALE_DIM, npadp);
    }

    // fp32 -> bf16 operand conversion (once, bandwidth-bound)
    #define CVT(src, dst, n) \
        cvt_f32_to_bf16<<<(unsigned)(((n)/4 + 255) / 256), 256, 0, stream>>>(src, dst, (long long)(n))
    CVT(features, feat_bf, (long long)B_ROWS * INPUT_DIM);
    CVT(W1,       W1_bf,   (long long)BELLY_DIM * INPUT_DIM);
    CVT(W2,       W2_bf,   (long long)SCALE_DIM * BELLY_DIM);
    CVT(Wt,       Wt_bf,   (long long)NUM_BINS * SCALE_DIM);
    CVT(Wp,       Wp_bf,   (long long)TOTAL_PAT * SCALE_DIM);
    #undef CVT

    // h = relu(features @ W1^T + b1) -> bf16 only (h is internal)
    gemm_wmma_bf16<1, 0, 1><<<dim3(BELLY_DIM / 128, B_ROWS / 128), 256, 0, stream>>>(
        feat_bf, W1_bf, b1, nullptr, h_bf, B_ROWS, BELLY_DIM, INPUT_DIM);

    // zraw = h @ W2^T -> fp32 (normalization wants fp32 precision)
    gemm_wmma_bf16<0, 1, 0><<<dim3(SCALE_DIM / 128, B_ROWS / 128), 256, 0, stream>>>(
        h_bf, W2_bf, nullptr, zraw, nullptr, B_ROWS, SCALE_DIM, BELLY_DIM);

    // z = zraw/||zraw||; bf16 copy; shift; class; max_pos
    normalize_row<<<B_ROWS, 256, 0, stream>>>(
        zraw, f2p_w, f2p_b, timesteps, z_out, z_bf, shift, cls, cls_out_f, pmax);

    // timestep_logits = z @ Wt^T + bt (Wt padded to 128 rows)
    gemm_wmma_bf16<0, 1, 0><<<dim3(NUM_BINS_P / 128, B_ROWS / 128), 256, 0, stream>>>(
        z_bf, Wt_bf, bt, tl_out, nullptr, B_ROWS, NUM_BINS, SCALE_DIM);

    // pattern_logits = z @ Wp^T + bp (Wp padded to 1024 rows)
    gemm_wmma_bf16<0, 1, 0><<<dim3(TOTAL_PAT_P / 128, B_ROWS / 128), 256, 0, stream>>>(
        z_bf, Wp_bf, bp, pl_out, nullptr, B_ROWS, TOTAL_PAT, SCALE_DIM);

    cantor_kernel<<<(B_ROWS + 255) / 256, 256, 0, stream>>>(
        cls, shift, pmax, alpha, cant_out, B_ROWS);
}
